// SSAM_60266981097585
// MI455X (gfx1250) — matrix operand order repaired and checked by reading, not compiled, not run on hardware
//
#include <hip/hip_runtime.h>

// Problem constants (match reference): B=4, CH=128, MID=2*CH=256, L=4096
#define B_N   4
#define CH_N  128
#define MID_N 256
#define L_SEQ 4096

typedef __attribute__((ext_vector_type(16))) __bf16 v16bf;
typedef __attribute__((ext_vector_type(8)))  float  v8f;

union AF { v16bf v; unsigned int w[8]; };

static __device__ __forceinline__ unsigned short f2bf(float f) {
  unsigned int u = __float_as_uint(f);
  u += 0x7FFFu + ((u >> 16) & 1u);           // round-to-nearest-even
  return (unsigned short)(u >> 16);
}

static __device__ __forceinline__ v8f wmma_bf16(v16bf a, v16bf b, v8f c) {
  // D(f32,16x16) = A(bf16,16x32) * B(bf16,32x16) + C
  return __builtin_amdgcn_wmma_f32_16x16x32_bf16(false, a, false, b, (short)0, c,
                                                 false, false);
}

// 16-bit A-matrix 16x32 layout (CDNA5 ISA): VGPR v, lane-half h ->
// K base: v<4 ? h*8 + v*2 : 16 + h*8 + (v-4)*2   (each VGPR = K,K+1 pair)
static __device__ __forceinline__ int a_k0(int v, int half) {
  return ((v & 4) << 2) + (half << 3) + ((v & 3) << 1);
}

// Async global->LDS 16-byte copy (ASYNCcnt-tracked DMA path, ISA 15.18.3
// op 98). LDS generic address: low 32 bits are the raw LDS byte offset.
static __device__ __forceinline__ void async_ld16(const void* g, void* lds) {
  const unsigned int l32 = (unsigned int)(unsigned long long)lds;
  asm volatile("global_load_async_to_lds_b128 %0, %1, off"
               :: "v"(l32), "v"(g) : "memory");
}
static __device__ __forceinline__ void wait_async() {
  asm volatile("s_wait_asynccnt 0" ::: "memory");
}

// ---------------------------------------------------------------------------
// Kernel 1: 1x1-conv projections. q is written TRANSPOSED (qT[b][l][c]) so the
// attention kernels can stage Q tiles with contiguous async B128 copies;
// k[b][c][m] and v[b][c][l] are already contiguous the way their WMMA
// fragments need them. Tiny FLOP share -> plain VALU with wave-uniform weights.
// ---------------------------------------------------------------------------
__global__ void __launch_bounds__(256) qkv_proj_kernel(
    const float* __restrict__ x, const float* __restrict__ Wq,
    const float* __restrict__ Wk, const float* __restrict__ Wv,
    unsigned short* __restrict__ qTw, unsigned short* __restrict__ kw,
    unsigned short* __restrict__ vw)
{
  __shared__ float xs[CH_N][32];
  const int tid = threadIdx.x;
  const int b  = blockIdx.x >> 7;          // 128 l-tiles per batch
  const int l0 = (blockIdx.x & 127) * 32;
  const float* xb = x + (size_t)b * CH_N * L_SEQ;

  for (int i = tid; i < CH_N * 32; i += 256)
    xs[i >> 5][i & 31] = xb[(size_t)(i >> 5) * L_SEQ + l0 + (i & 31)];
  __syncthreads();

  const int l  = tid & 31;
  const int og = tid >> 5;                 // wave-uniform output group
  for (int oo = 0; oo < 80; ++oo) {
    const int o = og + oo * 8;             // 0..639: 256 q | 256 k | 128 v
    const float* wrow;
    if (o < 256)      wrow = Wq + o * CH_N;
    else if (o < 512) wrow = Wk + (o - 256) * CH_N;
    else              wrow = Wv + (o - 512) * CH_N;
    float acc = 0.f;
    #pragma unroll 16
    for (int c = 0; c < CH_N; ++c) acc = fmaf(wrow[c], xs[c][l], acc);
    const unsigned short r = f2bf(acc);
    if (o < 256)
      qTw[((size_t)b * L_SEQ + l0 + l) * MID_N + o] = r;           // qT scatter
    else if (o < 512)
      kw[((size_t)b * MID_N + (o - 256)) * L_SEQ + l0 + l] = r;
    else
      vw[((size_t)b * CH_N + (o - 512)) * L_SEQ + l0 + l] = r;
  }
}

// ---------------------------------------------------------------------------
// Kernel 2: softmax statistics M[l], Zinv[l]. Double-buffered async K tiles:
// issue tile t+1, compute WMMA on tile t, one barrier per iteration.
// Score accumulation split into 4 independent WMMA chains (K-step parity x
// two N-tiles) so the scheduler can fill the bf16 WMMA->WMMA hazard window
// (5 slots, ISA 7.12.1) with real work instead of v_nop.
// ---------------------------------------------------------------------------
__global__ void __launch_bounds__(256) softmax_stats_kernel(
    const unsigned short* __restrict__ qTg, const unsigned short* __restrict__ kg,
    float* __restrict__ Mg, float* __restrict__ Zg)
{
  __shared__ union {
    unsigned short Qs[64][264];     // 64 l x 256 c (+8 pad)
    unsigned short Ks[2][256][40];  // ping-pong 256 c x 32 m (+8 pad)
  } sm;

  const int tid  = threadIdx.x;
  const int b    = blockIdx.y;
  const int l0   = blockIdx.x * 64;
  const int wv   = tid >> 5, lane = tid & 31;
  const int half = lane >> 4, l16 = lane & 15;
  const unsigned short* qTb = qTg + (size_t)b * L_SEQ * MID_N;
  const unsigned short* kb  = kg  + (size_t)b * MID_N * L_SEQ;

  // --- async-stage Q tile: 64 rows x 512B, 2048 x 16B chunks ---
  #pragma unroll
  for (int r = 0; r < 8; ++r) {
    const int cid = tid + r * 256;
    const int l = cid >> 5, j = cid & 31;
    async_ld16(qTb + (size_t)(l0 + l) * MID_N + j * 8, &sm.Qs[l][j * 8]);
  }
  wait_async();
  __syncthreads();

  AF qa[8];                                // A frags: 16 l x 256 c, resident
  const int qrow = wv * 16 + l16;
  #pragma unroll
  for (int kk = 0; kk < 8; ++kk)
    #pragma unroll
    for (int v = 0; v < 8; ++v)
      qa[kk].w[v] = *(const unsigned int*)&sm.Qs[qrow][kk * 32 + a_k0(v, half)];
  __syncthreads();                         // Qs dead; Ks may overwrite union

  // --- prime K pipeline: tile 0 -> buffer 0 (1024 x 16B chunks) ---
  #pragma unroll
  for (int r = 0; r < 4; ++r) {
    const int cid = tid + r * 256;
    const int c = cid >> 2, j = cid & 3;
    async_ld16(kb + (size_t)c * L_SEQ + j * 8, &sm.Ks[0][c][j * 8]);
  }

  float mrow[8], lsum[8];
  #pragma unroll
  for (int r = 0; r < 8; ++r) { mrow[r] = -1e30f; lsum[r] = 0.f; }

  for (int mt = 0; mt < L_SEQ / 32; ++mt) {
    wait_async();
    __syncthreads();
    const int buf = mt & 1;
    if (mt + 1 < L_SEQ / 32) {             // overlap next tile with compute
      const int m0n = (mt + 1) * 32;
      #pragma unroll
      for (int r = 0; r < 4; ++r) {
        const int cid = tid + r * 256;
        const int c = cid >> 2, j = cid & 3;
        async_ld16(kb + (size_t)c * L_SEQ + m0n + j * 8,
                   &sm.Ks[buf ^ 1][c][j * 8]);
      }
    }

    v8f s0c[2], s1c[2];
    s0c[0] = (v8f){}; s0c[1] = (v8f){}; s1c[0] = (v8f){}; s1c[1] = (v8f){};
    #pragma unroll
    for (int kk = 0; kk < 8; ++kk) {       // K dim = 256 channels
      AF b0, b1;                           // B layout: lane<->N(m), slot<->K(c)
      const unsigned short* krow = &sm.Ks[buf][kk * 32][l16];
      #pragma unroll
      for (int j = 0; j < 8; ++j) {
        const int k0 = a_k0(j, half);
        b0.w[j] = (unsigned int)krow[k0 * 40] | ((unsigned int)krow[(k0 + 1) * 40] << 16);
        b1.w[j] = (unsigned int)krow[k0 * 40 + 16] | ((unsigned int)krow[(k0 + 1) * 40 + 16] << 16);
      }
      const int p = kk & 1;                // 4 independent accumulator chains
      s0c[p] = wmma_bf16(qa[kk].v, b0.v, s0c[p]);
      s1c[p] = wmma_bf16(qa[kk].v, b1.v, s1c[p]);
    }
    const v8f s0 = s0c[0] + s0c[1];
    const v8f s1 = s1c[0] + s1c[1];

    // Online max/sum per row (row = VGPR r + lane-half; 16 lanes per row)
    #pragma unroll
    for (int r = 0; r < 8; ++r) {
      float tm = fmaxf(s0[r], s1[r]);
      #pragma unroll
      for (int off = 1; off < 16; off <<= 1) tm = fmaxf(tm, __shfl_xor(tm, off, 32));
      const float mnew = fmaxf(mrow[r], tm);
      const float sc   = __expf(mrow[r] - mnew);
      float rs = __expf(s0[r] - mnew) + __expf(s1[r] - mnew);
      #pragma unroll
      for (int off = 1; off < 16; off <<= 1) rs += __shfl_xor(rs, off, 32);
      lsum[r] = lsum[r] * sc + rs;
      mrow[r] = mnew;
    }
  }

  if (l16 == 0) {                          // one writer per row
    #pragma unroll
    for (int r = 0; r < 8; ++r) {
      const int l = l0 + wv * 16 + r + 8 * half;   // D layout: M = r + 8*half
      Mg[(size_t)b * L_SEQ + l] = mrow[r];
      Zg[(size_t)b * L_SEQ + l] = 1.0f / lsum[r];
    }
  }
}

// ---------------------------------------------------------------------------
// Kernel 3: out[b,c,m] = sum_l v[b,c,l] * exp(S[l,m]-M[l]) * Zinv[l]
// K tile resident in LDS; per l-step: S via WMMA -> P bf16 via LDS -> V @ P.
// Async staging with lifetime-overlap: next-Q issued during the O-GEMM,
// next-V after Vls consumption; softmax stats double-buffered. S accumulated
// in 2 independent chains to fill the WMMA hazard window.
// ---------------------------------------------------------------------------
__global__ void __launch_bounds__(256) attn_out_kernel(
    const unsigned short* __restrict__ qTg, const unsigned short* __restrict__ kg,
    const unsigned short* __restrict__ vg, const float* __restrict__ Mg,
    const float* __restrict__ Zg, float* __restrict__ out)
{
  __shared__ unsigned short Kls[MID_N][40];  // 256 c x 32 m (resident)
  __shared__ unsigned short Qls[32][264];    // 32 l x 256 c (per l-step)
  __shared__ unsigned short Vls[CH_N][40];   // 128 c x 32 l (per l-step)
  __shared__ unsigned short Pls[32][40];     // 32 l x 32 m probabilities
  __shared__ float Mr[2][32], Zr[2][32];

  const int tid  = threadIdx.x;
  const int b    = blockIdx.y;
  const int m0   = blockIdx.x * 32;
  const int wv   = tid >> 5, lane = tid & 31;
  const int half = lane >> 4, l16 = lane & 15;
  const unsigned short* qTb = qTg + (size_t)b * L_SEQ * MID_N;
  const unsigned short* kb  = kg  + (size_t)b * MID_N * L_SEQ;
  const unsigned short* vb  = vg  + (size_t)b * CH_N * L_SEQ;

  // --- prologue: async K (resident), Q/V tile 0, stats tile 0 ---
  #pragma unroll
  for (int r = 0; r < 4; ++r) {            // Kls: 1024 x 16B
    const int cid = tid + r * 256;
    const int c = cid >> 2, j = cid & 3;
    async_ld16(kb + (size_t)c * L_SEQ + m0 + j * 8, &Kls[c][j * 8]);
  }
  #pragma unroll
  for (int r = 0; r < 4; ++r) {            // Qls tile 0: 1024 x 16B
    const int cid = tid + r * 256;
    const int l = cid >> 5, j = cid & 31;
    async_ld16(qTb + (size_t)l * MID_N + j * 8, &Qls[l][j * 8]);
  }
  #pragma unroll
  for (int r = 0; r < 2; ++r) {            // Vls tile 0: 512 x 16B
    const int cid = tid + r * 256;
    const int c = cid >> 2, j = cid & 3;
    async_ld16(vb + (size_t)c * L_SEQ + j * 8, &Vls[c][j * 8]);
  }
  if (tid < 32) {
    Mr[0][tid] = Mg[(size_t)b * L_SEQ + tid];
    Zr[0][tid] = Zg[(size_t)b * L_SEQ + tid];
  }
  wait_async();
  __syncthreads();

  const int lsub = wv >> 1, msub = wv & 1;   // wave -> S quadrant
  const int cw   = wv * 32;                  // wave -> 32 output channels
  v8f zz = {};
  v8f accO[2][2];
  accO[0][0] = zz; accO[0][1] = zz; accO[1][0] = zz; accO[1][1] = zz;

  for (int ls = 0; ls < L_SEQ / 32; ++ls) {
    const int pb = ls & 1;

    // --- S quadrant (16 l x 16 m), K-dim = 256 channels, 2 chains ---
    v8f sc2[2];
    sc2[0] = (v8f){}; sc2[1] = (v8f){};
    #pragma unroll
    for (int kk = 0; kk < 8; ++kk) {
      AF a, bbf;
      const int arow = lsub * 16 + l16;
      #pragma unroll
      for (int v = 0; v < 8; ++v)
        a.w[v] = *(const unsigned int*)&Qls[arow][kk * 32 + a_k0(v, half)];
      const unsigned short* krow = &Kls[kk * 32][msub * 16 + l16];
      #pragma unroll
      for (int j = 0; j < 8; ++j) { const int k0 = a_k0(j, half); bbf.w[j] = (unsigned int)krow[k0 * 40] | ((unsigned int)krow[(k0 + 1) * 40] << 16); }
      sc2[kk & 1] = wmma_bf16(a.v, bbf.v, sc2[kk & 1]);
    }
    const v8f s = sc2[0] + sc2[1];

    // --- P = exp(S - M[l]) * Zinv[l] -> bf16 LDS (D->B transpose) ---
    #pragma unroll
    for (int r = 0; r < 8; ++r) {
      const int lrow = lsub * 16 + r + 8 * half;           // D layout row
      const float p = __expf(s[r] - Mr[pb][lrow]) * Zr[pb][lrow];
      Pls[lrow][msub * 16 + l16] = f2bf(p);
    }
    __syncthreads();                       // Pls visible; Qls reads retired

    if (ls + 1 < L_SEQ / 32) {             // next-Q overlaps the O-GEMM
      const int lq0n = (ls + 1) * 32;
      #pragma unroll
      for (int r = 0; r < 4; ++r) {
        const int cid = tid + r * 256;
        const int l = cid >> 5, j = cid & 31;
        async_ld16(qTb + (size_t)(lq0n + l) * MID_N + j * 8, &Qls[l][j * 8]);
      }
      if (tid < 32) {
        Mr[pb ^ 1][tid] = Mg[(size_t)b * L_SEQ + lq0n + tid];
        Zr[pb ^ 1][tid] = Zg[(size_t)b * L_SEQ + lq0n + tid];
      }
    }

    // --- out += V(32c x 32l) @ P(32l x 32m); 4 independent accumulators ---
    AF bp0, bp1;                                           // lane <-> N(m)
    #pragma unroll
    for (int j = 0; j < 8; ++j) {
      const int k0 = a_k0(j, half);
      bp0.w[j] = (unsigned int)Pls[k0][l16] | ((unsigned int)Pls[k0 + 1][l16] << 16);
      bp1.w[j] = (unsigned int)Pls[k0][16 + l16] | ((unsigned int)Pls[k0 + 1][16 + l16] << 16);
    }
    #pragma unroll
    for (int ci = 0; ci < 2; ++ci) {
      AF av;                                               // A: M=c, K=l
      const int crow = cw + ci * 16 + l16;
      #pragma unroll
      for (int v = 0; v < 8; ++v)
        av.w[v] = *(const unsigned int*)&Vls[crow][a_k0(v, half)];
      accO[ci][0] = wmma_bf16(av.v, bp0.v, accO[ci][0]);
      accO[ci][1] = wmma_bf16(av.v, bp1.v, accO[ci][1]);
    }
    __syncthreads();                       // Vls/Pls reads retired

    if (ls + 1 < L_SEQ / 32) {             // next-V
      const int lq0n = (ls + 1) * 32;
      #pragma unroll
      for (int r = 0; r < 2; ++r) {
        const int cid = tid + r * 256;
        const int c = cid >> 2, j = cid & 3;
        async_ld16(vb + (size_t)c * L_SEQ + lq0n + j * 8, &Vls[c][j * 8]);
      }
    }
    wait_async();                          // Q (+V) for next step complete
    __syncthreads();
  }

  // Epilogue: D layout -> out[b][c][m]; lanes sweep m (coalesced)
  float* ob = out + (size_t)b * CH_N * L_SEQ;
  #pragma unroll
  for (int ci = 0; ci < 2; ++ci)
    #pragma unroll
    for (int r = 0; r < 8; ++r) {
      const int c = cw + ci * 16 + r + 8 * half;
      ob[(size_t)c * L_SEQ + m0 + l16]      = accO[ci][0][r];
      ob[(size_t)c * L_SEQ + m0 + 16 + l16] = accO[ci][1][r];
    }
}

// ---------------------------------------------------------------------------
extern "C" void kernel_launch(void* const* d_in, const int* in_sizes, int n_in,
                              void* d_out, int out_size, void* d_ws, size_t ws_size,
                              hipStream_t stream)
{
  (void)in_sizes; (void)n_in; (void)out_size; (void)ws_size;
  const float* x  = (const float*)d_in[0];
  const float* Wq = (const float*)d_in[1];
  const float* Wk = (const float*)d_in[2];
  const float* Wv = (const float*)d_in[3];
  float* out = (float*)d_out;

  // Workspace layout (bf16 qT/k/v + fp32 stats): ~20.1 MB total
  const size_t qElems = (size_t)B_N * MID_N * L_SEQ;   // 4,194,304
  const size_t vElems = (size_t)B_N * CH_N * L_SEQ;    // 2,097,152
  unsigned short* qTw = (unsigned short*)d_ws;
  unsigned short* kw  = qTw + qElems;
  unsigned short* vw  = kw + qElems;
  float* Mg = (float*)(vw + vElems);
  float* Zg = Mg + (size_t)B_N * L_SEQ;

  qkv_proj_kernel<<<dim3(B_N * (L_SEQ / 32)), 256, 0, stream>>>(x, Wq, Wk, Wv,
                                                                qTw, kw, vw);
  softmax_stats_kernel<<<dim3(L_SEQ / 64, B_N), 256, 0, stream>>>(qTw, kw, Mg, Zg);
  attn_out_kernel<<<dim3(L_SEQ / 32, B_N), 256, 0, stream>>>(qTw, kw, vw, Mg, Zg,
                                                             out);
}
